// InfiniTransformer_7645041787084
// MI455X (gfx1250) — matrix operand order repaired and checked by reading, not compiled, not run on hardware
//
#include <hip/hip_runtime.h>
#include <math.h>

// ---------------- config ----------------
#ifndef USE_ASYNC_LDS
#define USE_ASYNC_LDS 1   // gfx1250 GLOBAL_LOAD_ASYNC_TO_LDS_B128 via inline asm
#endif

// ---------------- types ----------------
typedef __attribute__((ext_vector_type(16))) __bf16 v16bf;
typedef __attribute__((ext_vector_type(8)))  float  v8f;
typedef __attribute__((ext_vector_type(2)))  float  v2f;

struct __attribute__((aligned(16))) U4 { unsigned int a, b, c, d; };
union BF16x16 { v16bf v; unsigned short u[16]; U4 q[2]; };

#if defined(__has_builtin)
#if __has_builtin(__builtin_amdgcn_wmma_f32_16x16x4_f32)
#define HAVE_WMMA_F32X4 1
#endif
#endif

// ---------------- problem constants ----------------
constexpr int kB = 2, kS = 8192, kD = 1024, kH = 8, kDK = 64, kDV = 64;
constexpr int kDH = 4096, kSEG = 2048, kNSEG = 4;
constexpr int kM = kB * kS;          // 16384 token rows
constexpr int kNQKV = 3 * kH * kDK;  // 1536

// ---------------- scalar helpers ----------------
__device__ __forceinline__ unsigned short f32_to_bf16(float f) {
  unsigned int u = __float_as_uint(f);
  unsigned int r = 0x7FFFu + ((u >> 16) & 1u);
  return (unsigned short)((u + r) >> 16);
}
__device__ __forceinline__ float bf16_to_f32(unsigned short u) {
  return __uint_as_float(((unsigned int)u) << 16);
}
__device__ __forceinline__ float elu_p1(float x) {  // elu(x)+1
  return x > 0.f ? x + 1.f : __expf(x);
}
__device__ __forceinline__ float gelu_exact(float x) {
  return 0.5f * x * (1.f + erff(x * 0.70710678118654752f));
}

// ---------------- gfx1250 async copy global->LDS ----------------
// 16B per lane, tracked by ASYNCcnt (CDNA5 ISA).
__device__ __forceinline__ void async_load_b128(unsigned int lds_byte_addr, const void* gptr) {
#if USE_ASYNC_LDS
  asm volatile("global_load_async_to_lds_b128 %0, %1, off"
               :: "v"(lds_byte_addr), "v"((unsigned long long)(size_t)gptr)
               : "memory");
#else
  (void)lds_byte_addr; (void)gptr;
#endif
}
__device__ __forceinline__ void wait_async0() {
#if USE_ASYNC_LDS
  asm volatile("s_wait_asynccnt 0" ::: "memory");
#endif
}
__device__ __forceinline__ unsigned int lds_addr_of(const void* p) {
  // generic LDS pointer: low 32 bits are the LDS byte offset (ISA §10.2 aperture rules)
  return (unsigned int)(size_t)p;
}

// A-matrix 16x32 bf16 fragment from LDS per ISA 7.12.2 layout.
// lanes 0-15: M=lane; lanes 16-31: M=lane-16 (K offset +8).
__device__ __forceinline__ v16bf ldsA_frag(const unsigned short* base, int lda, int row0, int lane) {
  BF16x16 r;
  const int m  = row0 + (lane & 15);
  const int ko = (lane & 16) ? 8 : 0;
#pragma unroll
  for (int e = 0; e < 8; ++e) {
    const int kb = ((e & 4) ? 16 : 0) + ko + ((e & 3) << 1);
    const unsigned int pr = *(const unsigned int*)(base + m * lda + kb);
    r.u[2 * e]     = (unsigned short)(pr & 0xffffu);
    r.u[2 * e + 1] = (unsigned short)(pr >> 16);
  }
  return r.v;
}

__device__ __forceinline__ v8f wmma_bf16(v16bf a, v16bf b, v8f c) {
  return __builtin_amdgcn_wmma_f32_16x16x32_bf16(false, a, false, b, (short)0, c, false, false);
}

// ---------------- cast / pack kernels ----------------
__global__ __launch_bounds__(256) void cast_bf16_kernel(const float* __restrict__ src,
                                                        unsigned short* __restrict__ dst,
                                                        long long n) {
  long long i = (long long)blockIdx.x * 256 + threadIdx.x;
  const long long stride = (long long)gridDim.x * 256;
  for (; i < n; i += stride) dst[i] = f32_to_bf16(src[i]);
}

__global__ __launch_bounds__(256) void pack_qkvw_kernel(const float* __restrict__ Wq,
                                                        const float* __restrict__ Wk,
                                                        const float* __restrict__ Wv,
                                                        unsigned short* __restrict__ dst) {
  const int n = kD * kNQKV;
  for (int i = blockIdx.x * 256 + threadIdx.x; i < n; i += gridDim.x * 256) {
    const int row = i / kNQKV, col = i % kNQKV;
    float v;
    if (col < 512)       v = Wq[row * 512 + col];
    else if (col < 1024) v = Wk[row * 512 + (col - 512)];
    else                 v = Wv[row * 512 + (col - 1024)];
    dst[i] = f32_to_bf16(v);
  }
}

// ---------------- WMMA GEMM: C[MxN] = A[MxK](bf16) x B[KxN](bf16) ----------------
// Double-buffered LDS, async global->LDS staging; one barrier per K-step.
// mode: 1 = store bf16; 2 = +bias, GELU, store bf16; 3 = +bias +resid, store f32; 0 = store f32
__global__ __launch_bounds__(256) void gemm_bf16_wmma(
    const unsigned short* __restrict__ A, const unsigned short* __restrict__ Bw,
    const float* __restrict__ bias, const float* __restrict__ resid,
    void* __restrict__ outp, int M, int N, int K, int mode) {
  __shared__ __align__(16) unsigned short As[2][128][32];
  __shared__ __align__(16) unsigned short Bs[2][32][128];

  const int tid = threadIdx.x, lane = tid & 31, wave = tid >> 5;
  const int half = lane >> 4, col15 = lane & 15;
  const long long bm = (long long)blockIdx.x * 128;
  const long long bn = (long long)blockIdx.y * 128;
  const int wm = (wave & 3) * 32, wn = (wave >> 2) * 64;

  const v8f vz8 = {0.f, 0.f, 0.f, 0.f, 0.f, 0.f, 0.f, 0.f};
  v8f acc[2][4];
#pragma unroll
  for (int mi = 0; mi < 2; ++mi)
#pragma unroll
    for (int ni = 0; ni < 4; ++ni) acc[mi][ni] = vz8;

  const int ar = tid >> 1, ac = (tid & 1) * 16;   // A tile: 128x32, 32B/thread
  const int br = tid >> 3, bc = (tid & 7) * 16;   // B tile: 32x128, 32B/thread

  auto stage = [&](int buf, int kt) {
    const unsigned short* ga = A + (bm + ar) * (long long)K + kt + ac;
    const unsigned short* gb = Bw + (long long)(kt + br) * N + bn + bc;
#if USE_ASYNC_LDS
    const unsigned int la = lds_addr_of(&As[buf][ar][ac]);
    const unsigned int lb = lds_addr_of(&Bs[buf][br][bc]);
    async_load_b128(la,      ga);
    async_load_b128(la + 16, ga + 8);
    async_load_b128(lb,      gb);
    async_load_b128(lb + 16, gb + 8);
#else
    U4* d4 = (U4*)&As[buf][ar][ac];
    const U4* s4 = (const U4*)ga;
    d4[0] = s4[0]; d4[1] = s4[1];
    U4* d4b = (U4*)&Bs[buf][br][bc];
    const U4* s4b = (const U4*)gb;
    d4b[0] = s4b[0]; d4b[1] = s4b[1];
#endif
  };

  stage(0, 0);                       // prologue: tile 0 in flight
  int buf = 0;
  for (int kt = 0; kt < K; kt += 32, buf ^= 1) {
    wait_async0();                   // my async writes for buf are done
    __syncthreads();                 // everyone's writes visible; prior reads of buf^1 done
    if (kt + 32 < K) stage(buf ^ 1, kt + 32);  // next tile loads overlap this tile's WMMAs

    v16bf af[2];
    af[0] = ldsA_frag(&As[buf][0][0], 32, wm, lane);
    af[1] = ldsA_frag(&As[buf][0][0], 32, wm + 16, lane);
    BF16x16 bf_[4];
#pragma unroll
    for (int ni = 0; ni < 4; ++ni) {  // B frag: N=lane%16, K per element in table order
      for (int s = 0; s < 16; ++s) {
        const int k = ((s & 8) ? 16 : 0) + 8 * half + (s & 7);
        bf_[ni].u[s] = Bs[buf][k][wn + ni * 16 + col15];
      }
    }
#pragma unroll
    for (int mi = 0; mi < 2; ++mi)
#pragma unroll
      for (int ni = 0; ni < 4; ++ni)
        acc[mi][ni] = wmma_bf16(af[mi], bf_[ni].v, acc[mi][ni]);
  }

#pragma unroll
  for (int mi = 0; mi < 2; ++mi)
#pragma unroll
    for (int ni = 0; ni < 4; ++ni)
#pragma unroll
      for (int i = 0; i < 8; ++i) {
        const long long row = bm + wm + mi * 16 + i + 8 * half;
        const long long col = bn + wn + ni * 16 + col15;
        float v = acc[mi][ni][i];
        if (bias) v += bias[col];
        if (mode == 2) v = gelu_exact(v);
        if (mode == 3) v += resid[row * N + col];
        if (mode == 1 || mode == 2)
          ((unsigned short*)outp)[row * N + col] = f32_to_bf16(v);
        else
          ((float*)outp)[row * N + col] = v;
      }
}

// ---------------- compressive-memory attention ----------------
// grid = B*H workgroups, 256 threads (8 waves). Sequential over NSEG segments:
//   phase 1: flash attention + memory retrieval per 16-row query tile (WMMA)
//   phase 2: mem += sk^T v ; z += sum(sk) cooperative update in fp32 LDS
__global__ __launch_bounds__(256) void attn_kernel(
    const unsigned short* __restrict__ qkv,  // [kM][1536] bf16 (q|k|v per head-major)
    const float* __restrict__ betas,         // [kH*kDV]
    unsigned short* __restrict__ att)        // [kM][512] bf16
{
  constexpr int MEMF_OFF = 0;                        // 64x64 f32      (16384 B)
  constexpr int Z_OFF    = 64 * 64 * 4;              // 64 f32         (256 B)
  constexpr int KST_OFF  = Z_OFF + 64 * 4;           // 8 waves x 32x64 bf16 (32768 B)
  constexpr int PST_OFF  = KST_OFF + 8 * 32 * 64 * 2;// 8 waves x 16x32 bf16 (8192 B)
  __shared__ __align__(16) char smem[PST_OFF + 8 * 16 * 32 * 2];  // 57600 B

  float* memf = (float*)(smem + MEMF_OFF);
  float* zf   = (float*)(smem + Z_OFF);

  const int tid = threadIdx.x;
  const int wave = tid >> 5, lane = tid & 31;
  const int half = lane >> 4, col15 = lane & 15;
  const int bidx = blockIdx.x / kH, h = blockIdx.x % kH;

  unsigned short* kst_w = (unsigned short*)(smem + KST_OFF) + wave * (32 * 64);
  unsigned short* pst_w = (unsigned short*)(smem + PST_OFF) + wave * (16 * 32);

  for (int i = tid; i < 64 * 64; i += 256) memf[i] = 0.f;
  if (tid < 64) zf[tid] = 1.0f / (float)kDK;
  __syncthreads();

  float gates[4];
#pragma unroll
  for (int ni = 0; ni < 4; ++ni) {
    const float bv = betas[h * 64 + ni * 16 + col15];
    gates[ni] = 1.f / (1.f + __expf(-bv));
  }
  const float scale = 0.125f;  // 1/sqrt(DK)
  const v8f vz8 = {0.f, 0.f, 0.f, 0.f, 0.f, 0.f, 0.f, 0.f};

  for (int seg = 0; seg < kNSEG; ++seg) {
    const long long srow0 = (long long)bidx * kS + (long long)seg * kSEG;

    // ---------- phase 1: attention (uses mem/z from previous segments) ----------
    for (int qt = wave; qt < kSEG / 16; qt += 8) {
      BF16x16 qa[2];  // q A-frags, 16x64 over two K=32 steps
      {
        const unsigned short* qbase = qkv + (srow0 + qt * 16 + col15) * 1536 + h * 64;
        const int ko = half ? 8 : 0;
#pragma unroll
        for (int f = 0; f < 2; ++f)
#pragma unroll
          for (int e = 0; e < 8; ++e) {
            const int kb = f * 32 + ((e & 4) ? 16 : 0) + ko + ((e & 3) << 1);
            const unsigned int pr = *(const unsigned int*)(qbase + kb);
            qa[f].u[2 * e]     = (unsigned short)(pr & 0xffffu);
            qa[f].u[2 * e + 1] = (unsigned short)(pr >> 16);
          }
      }
      v8f cd[4] = {vz8, vz8, vz8, vz8};  // unnormalized O (16x64)
      float mrow[8], lrow[8];
#pragma unroll
      for (int i = 0; i < 8; ++i) { mrow[i] = -1e30f; lrow[i] = 0.f; }

      const int kend = qt * 16 + 16;  // causal limit (exclusive)
      for (int kc = 0; kc < kend; kc += 32) {
        {  // stage K chunk (32 keys x dims 0..31) into per-wave LDS; lane=key row
          const unsigned short* gk = qkv + (srow0 + kc + lane) * 1536 + 512 + h * 64;
#if USE_ASYNC_LDS
          const unsigned int l0 = lds_addr_of(kst_w + lane * 64);
          async_load_b128(l0,      gk);
          async_load_b128(l0 + 16, gk + 8);
          async_load_b128(l0 + 32, gk + 16);
          async_load_b128(l0 + 48, gk + 24);
          wait_async0();              // wave-local: data visible to this wave's ds reads
#else
          const U4* s4 = (const U4*)gk;
          U4* d4 = (U4*)(kst_w + lane * 64);
          d4[0] = s4[0]; d4[1] = s4[1]; d4[2] = s4[2]; d4[3] = s4[3];
#endif
        }
        v8f s0 = vz8, s1 = vz8;  // two 16x16 score tiles
#pragma unroll
        for (int kk = 0; kk < 64; kk += 32) {
          BF16x16 b0, b1;  // B = k^T frag: N = key (lane%16), K-dim = d (elem, table order)
#pragma unroll
          for (int e = 0; e < 16; ++e) {
            const int k = ((e & 8) ? 16 : 0) + 8 * half + (e & 7);
            b0.u[e] = kst_w[col15 * 64 + kk + k];
            b1.u[e] = kst_w[(16 + col15) * 64 + kk + k];
          }
          s0 = wmma_bf16(qa[kk >> 5].v, b0.v, s0);
          s1 = wmma_bf16(qa[kk >> 5].v, b1.v, s1);
        }
        const int rowb = qt * 16 + 8 * half;
        float p0[8], p1[8];
#pragma unroll
        for (int i = 0; i < 8; ++i) {
          const int row = rowb + i;
          float a0 = s0[i] * scale; if (kc + col15 > row)       a0 = -1e30f;
          float a1 = s1[i] * scale; if (kc + 16 + col15 > row)  a1 = -1e30f;
          float mx = fmaxf(a0, a1);
          mx = fmaxf(mx, __shfl_xor(mx, 1));
          mx = fmaxf(mx, __shfl_xor(mx, 2));
          mx = fmaxf(mx, __shfl_xor(mx, 4));
          mx = fmaxf(mx, __shfl_xor(mx, 8));
          const float mnew  = fmaxf(mrow[i], mx);
          const float alpha = __expf(mrow[i] - mnew);
          mrow[i] = mnew;
          const float e0 = __expf(a0 - mnew), e1 = __expf(a1 - mnew);
          float rs = e0 + e1;
          rs += __shfl_xor(rs, 1); rs += __shfl_xor(rs, 2);
          rs += __shfl_xor(rs, 4); rs += __shfl_xor(rs, 8);
          lrow[i] = lrow[i] * alpha + rs;
          p0[i] = e0; p1[i] = e1;
#pragma unroll
          for (int ni = 0; ni < 4; ++ni) cd[ni][i] *= alpha;
        }
        // transpose P (C-layout) -> A-layout through per-wave LDS
#pragma unroll
        for (int i = 0; i < 8; ++i) {
          pst_w[(i + 8 * half) * 32 + col15]      = f32_to_bf16(p0[i]);
          pst_w[(i + 8 * half) * 32 + 16 + col15] = f32_to_bf16(p1[i]);
        }
        const v16bf pa = ldsA_frag(pst_w, 32, 0, lane);
        const unsigned short* vbase = qkv + (srow0 + kc) * 1536 + 1024 + h * 64;
#pragma unroll
        for (int ni = 0; ni < 4; ++ni) {  // B = V frag: N=lane%16, K=key per element (table order)
          BF16x16 bv;
          for (int e = 0; e < 16; ++e) {
            const int k = ((e & 8) ? 16 : 0) + 8 * half + (e & 7);
            bv.u[e] = vbase[k * 1536 + ni * 16 + col15];
          }
          cd[ni] = wmma_bf16(pa, bv.v, cd[ni]);
        }
      }

      // ----- compressive memory retrieval: sq@mem / sq@z -----
      float* sqt = (float*)kst_w;  // reuse per-wave stage as 16x64 f32 sq tile
      {
        const int ko = half ? 8 : 0;
#pragma unroll
        for (int f = 0; f < 2; ++f)
#pragma unroll
          for (int e = 0; e < 8; ++e) {
            const int kb = f * 32 + ((e & 4) ? 16 : 0) + ko + ((e & 3) << 1);
            sqt[col15 * 64 + kb]     = elu_p1(bf16_to_f32(qa[f].u[2 * e]));
            sqt[col15 * 64 + kb + 1] = elu_p1(bf16_to_f32(qa[f].u[2 * e + 1]));
          }
      }
      float dnm;
      {
        float part = 0.f;
        const int dbase = half * 32;
#pragma unroll 8
        for (int d = 0; d < 32; ++d) part += sqt[col15 * 64 + dbase + d] * zf[dbase + d];
        part += __shfl_xor(part, 16);
        dnm = part;  // denom of row col15, replicated in both halves
      }
      v8f cm[4] = {vz8, vz8, vz8, vz8};
#if defined(HAVE_WMMA_F32X4)
      for (int ks = 0; ks < 16; ++ks) {  // fp32 WMMA against fp32 memory matrix
        v2f a;
        const int kb = ks * 4 + (half ? 2 : 0);
        a[0] = sqt[col15 * 64 + kb];
        a[1] = sqt[col15 * 64 + kb + 1];
#pragma unroll
        for (int ni = 0; ni < 4; ++ni) {
          v2f bb;
          bb[0] = memf[kb * 64 + ni * 16 + col15];
          bb[1] = memf[(kb + 1) * 64 + ni * 16 + col15];
          cm[ni] = __builtin_amdgcn_wmma_f32_16x16x4_f32(false, a, false, bb,
                                                         (short)0, cm[ni], false, false);
        }
      }
#else
#pragma unroll
      for (int ni = 0; ni < 4; ++ni)
#pragma unroll
        for (int i = 0; i < 8; ++i) {
          float s = 0.f;
          const int row = i + 8 * half;
          for (int d = 0; d < 64; ++d) s += sqt[row * 64 + d] * memf[d * 64 + ni * 16 + col15];
          cm[ni][i] = s;
        }
#endif
      // combine gate*att_mem + (1-gate)*att_dot and store bf16
#pragma unroll
      for (int ni = 0; ni < 4; ++ni)
#pragma unroll
        for (int i = 0; i < 8; ++i) {
          const float dv = __shfl(dnm, i + 8 * half);
          const float am = cm[ni][i] / dv;
          const float ad = cd[ni][i] / lrow[i];
          const float o  = gates[ni] * am + (1.f - gates[ni]) * ad;
          const long long orow = srow0 + qt * 16 + i + 8 * half;
          att[orow * 512 + h * 64 + ni * 16 + col15] = f32_to_bf16(o);
        }
    }
    __syncthreads();

    // ---------- phase 2: mem += sk^T v ; z += sum_s sk ----------
    {
      float* skst = (float*)(smem + KST_OFF);  // [32][64] (aliases kstage)
      float* vst  = skst + 32 * 64;            // [32][64]
      float acc[16];
#pragma unroll
      for (int j = 0; j < 16; ++j) acc[j] = 0.f;
      float zacc = 0.f;
      const int d0 = tid >> 2;          // 0..63
      const int v0 = (tid & 3) << 4;    // 0,16,32,48
      for (int kc = 0; kc < kSEG; kc += 32) {
        __syncthreads();
        for (int t = tid; t < 32 * 64; t += 256) {
          const int s_ = t >> 6, d_ = t & 63;
          const long long base = (srow0 + kc + s_) * 1536 + h * 64;
          skst[t] = elu_p1(bf16_to_f32(qkv[base + 512 + d_]));
          vst[t]  = bf16_to_f32(qkv[base + 1024 + d_]);
        }
        __syncthreads();
        for (int s_ = 0; s_ < 32; ++s_) {
          const float skv = skst[s_ * 64 + d0];
          if ((tid & 3) == 0) zacc += skv;
#pragma unroll
          for (int j = 0; j < 16; ++j) acc[j] += skv * vst[s_ * 64 + v0 + j];
        }
      }
      __syncthreads();
#pragma unroll
      for (int j = 0; j < 16; ++j) memf[d0 * 64 + v0 + j] += acc[j];
      if ((tid & 3) == 0) zf[d0] += zacc;
      __syncthreads();
    }
  }
}

// ---------------- LayerNorm ----------------
__global__ __launch_bounds__(256) void ln_kernel(const float* __restrict__ y,
                                                 const float* __restrict__ g,
                                                 const float* __restrict__ bb,
                                                 float* __restrict__ out) {
  __shared__ float red[256];
  const long long row = blockIdx.x;
  const int tid = threadIdx.x;
  const float* yr = y + row * kD;
  float s = 0.f;
  for (int i = tid; i < kD; i += 256) s += yr[i];
  red[tid] = s; __syncthreads();
  for (int o = 128; o > 0; o >>= 1) { if (tid < o) red[tid] += red[tid + o]; __syncthreads(); }
  const float mu = red[0] * (1.0f / kD);
  __syncthreads();
  float v = 0.f;
  for (int i = tid; i < kD; i += 256) { const float d = yr[i] - mu; v += d * d; }
  red[tid] = v; __syncthreads();
  for (int o = 128; o > 0; o >>= 1) { if (tid < o) red[tid] += red[tid + o]; __syncthreads(); }
  const float inv = rsqrtf(red[0] * (1.0f / kD) + 1e-5f);
  for (int i = tid; i < kD; i += 256)
    out[row * kD + i] = (yr[i] - mu) * inv * g[i] + bb[i];
}

// ---------------- launch ----------------
extern "C" void kernel_launch(void* const* d_in, const int* in_sizes, int n_in,
                              void* d_out, int out_size, void* d_ws, size_t ws_size,
                              hipStream_t stream) {
  const float* x    = (const float*)d_in[0];
  const float* Wq   = (const float*)d_in[1];
  const float* Wk   = (const float*)d_in[2];
  const float* Wv   = (const float*)d_in[3];
  const float* Wo   = (const float*)d_in[4];
  const float* bet  = (const float*)d_in[5];
  const float* W1   = (const float*)d_in[6];
  const float* b1   = (const float*)d_in[7];
  const float* W2   = (const float*)d_in[8];
  const float* b2   = (const float*)d_in[9];
  const float* ln_g = (const float*)d_in[10];
  const float* ln_b = (const float*)d_in[11];

  size_t off = 0;
  auto alloc = [&](size_t bytes) {
    void* p = (char*)d_ws + off;
    off = (off + bytes + 255) & ~(size_t)255;
    return p;
  };
  unsigned short* xb    = (unsigned short*)alloc((size_t)kM * kD * 2);
  unsigned short* wqkvb = (unsigned short*)alloc((size_t)kD * kNQKV * 2);
  unsigned short* wob   = (unsigned short*)alloc((size_t)512 * kD * 2);
  unsigned short* w1b   = (unsigned short*)alloc((size_t)kD * kDH * 2);
  unsigned short* w2b   = (unsigned short*)alloc((size_t)kDH * kD * 2);
  unsigned short* qkvb  = (unsigned short*)alloc((size_t)kM * kNQKV * 2);
  unsigned short* attb  = (unsigned short*)alloc((size_t)kM * 512 * 2);
  unsigned short* ab    = (unsigned short*)alloc((size_t)kM * kD * 2);
  unsigned short* h1b   = (unsigned short*)alloc((size_t)kM * kDH * 2);
  float*          yb    = (float*)alloc((size_t)kM * kD * 4);
  (void)ws_size; (void)in_sizes; (void)n_in; (void)out_size;

  // casts / weight packing
  cast_bf16_kernel<<<2048, 256, 0, stream>>>(x,  xb,  (long long)kM * kD);
  pack_qkvw_kernel<<<1024, 256, 0, stream>>>(Wq, Wk, Wv, wqkvb);
  cast_bf16_kernel<<<512,  256, 0, stream>>>(Wo, wob, (long long)512 * kD);
  cast_bf16_kernel<<<2048, 256, 0, stream>>>(W1, w1b, (long long)kD * kDH);
  cast_bf16_kernel<<<2048, 256, 0, stream>>>(W2, w2b, (long long)kDH * kD);

  // QKV projection (fused q|k|v): [16384,1024] x [1024,1536] -> bf16
  gemm_bf16_wmma<<<dim3(kM / 128, kNQKV / 128), 256, 0, stream>>>(
      xb, wqkvb, nullptr, nullptr, qkvb, kM, kNQKV, kD, 1);

  // compressive-memory attention (per batch-head workgroup)
  attn_kernel<<<kB * kH, 256, 0, stream>>>(qkvb, bet, attb);

  // output projection: [16384,512] x [512,1024] -> bf16
  gemm_bf16_wmma<<<dim3(kM / 128, kD / 128), 256, 0, stream>>>(
      attb, wob, nullptr, nullptr, ab, kM, kD, 512, 1);

  // MLP up + GELU: [16384,1024] x [1024,4096] + b1 -> gelu -> bf16
  gemm_bf16_wmma<<<dim3(kM / 128, kDH / 128), 256, 0, stream>>>(
      ab, w1b, b1, nullptr, h1b, kM, kDH, kD, 2);

  // MLP down + bias + residual x: -> f32
  gemm_bf16_wmma<<<dim3(kM / 128, kD / 128), 256, 0, stream>>>(
      h1b, w2b, b2, x, yb, kM, kD, kDH, 3);

  // LayerNorm -> output
  ln_kernel<<<kM, 256, 0, stream>>>(yb, ln_g, ln_b, (float*)d_out);
}
